// QuantConv_86517821216413
// MI455X (gfx1250) — matrix operand order repaired and checked by reading, not compiled, not run on hardware
//
#include <hip/hip_runtime.h>

// ---------------------------------------------------------------------------
// QuantConv (6-bit weight fake-quant) + BN + QuantReLU for MI455X (gfx1250)
// Implicit-GEMM, V_WMMA_F32_16X16X32_F16, double-buffered LDS, async-to-LDS B.
// ---------------------------------------------------------------------------

typedef _Float16 half8 __attribute__((ext_vector_type(8)));
typedef _Float16 v16h  __attribute__((ext_vector_type(16)));
typedef float    v8f   __attribute__((ext_vector_type(8)));
typedef int      v4i   __attribute__((ext_vector_type(4)));

#define CIN    128
#define COUT   256
#define HWD    80
#define IMG    6400
#define MTOT   102400
#define WNELE  294912            // 256*128*9
#define MT     64                // workgroup M tile (pixels)
#define NT     256               // workgroup N tile (all output channels)
#define KC     32                // K chunk per WMMA step
#define KCP    48                // padded LDS row (96B: keeps b128 align, kills conflicts)
#define NSTEP  36                // 9 taps * 4 channel chunks

#if __has_builtin(__builtin_amdgcn_global_load_async_to_lds_b128)
#define USE_ASYNC 1
typedef __attribute__((address_space(1))) v4i gv4i;   // global int4
typedef __attribute__((address_space(3))) v4i lv4i;   // LDS int4
#else
#define USE_ASYNC 0
#endif

__device__ __forceinline__ void async_wait0() {
#if __has_builtin(__builtin_amdgcn_s_wait_asynccnt)
    __builtin_amdgcn_s_wait_asynccnt(0);
#else
    asm volatile("s_wait_asynccnt 0x0" ::: "memory");
#endif
}

// ---------------------------------------------------------------------------
// Kernel 1a/1b: absmax(|W|) via bitwise atomicMax (floats >= 0 order as uints)
// ---------------------------------------------------------------------------
__global__ void k_init(unsigned* __restrict__ ws) { ws[0] = 0u; }

__global__ void k_absmax(const float* __restrict__ w, unsigned* __restrict__ ws) {
    __shared__ float red[256];
    const int t = threadIdx.x;
    float m = 0.0f;
    for (int i = blockIdx.x * 256 + t; i < WNELE; i += gridDim.x * 256)
        m = fmaxf(m, fabsf(w[i]));
    red[t] = m;
    __syncthreads();
    for (int s = 128; s > 0; s >>= 1) {
        if (t < s) red[t] = fmaxf(red[t], red[t + s]);
        __syncthreads();
    }
    if (t == 0) atomicMax(ws, __float_as_uint(red[0]));
}

// ---------------------------------------------------------------------------
// Kernel 2: quantize weights (integers, exact in f16), relayout OIHW->[tap][O][C]
// so GEMM B rows are contiguous 64B runs of K. Also publish w_scale to ws[1].
// ---------------------------------------------------------------------------
__global__ void k_wquant(const float* __restrict__ w, const unsigned* __restrict__ wsu,
                         float* __restrict__ wsf, _Float16* __restrict__ wt2) {
    const int i = blockIdx.x * 256 + threadIdx.x;
    const float scale = __uint_as_float(wsu[0]) / 31.0f;
    if (i == 0) wsf[1] = scale;
    if (i >= WNELE) return;
    const int o   = i / (CIN * 9);
    const int c   = (i / 9) % CIN;
    const int tap = i % 9;
    float q = rintf(w[i] / scale);
    q = fminf(fmaxf(q, -31.0f), 31.0f);
    wt2[((size_t)(tap * COUT + o)) * CIN + c] = (_Float16)q;
}

// ---------------------------------------------------------------------------
// Staging helpers
// ---------------------------------------------------------------------------
__device__ __forceinline__ void a_gather(const float* __restrict__ x, long xbase,
                                         int ph, int pw, int tap, int c0, int a_kg,
                                         float areg[8]) {
    const int r  = tap / 3, s = tap - 3 * r;
    const int hh = ph + r - 1, ww = pw + s - 1;
    const bool valid = (hh >= 0) & (hh < HWD) & (ww >= 0) & (ww < HWD);
    const int hc = hh < 0 ? 0 : (hh > HWD - 1 ? HWD - 1 : hh);
    const int wc = ww < 0 ? 0 : (ww > HWD - 1 ? HWD - 1 : ww);
    const long base = xbase + (long)(c0 + a_kg) * IMG + hc * HWD + wc;
#pragma unroll
    for (int i = 0; i < 8; ++i) {
        const float v = x[base + (long)i * IMG];   // always in-bounds
        areg[i] = valid ? v : 0.0f;                // cndmask, no divergence
    }
}

__device__ __forceinline__ void a_commit(_Float16 (*Abuf)[KCP], int a_m, int a_kg,
                                         const float areg[8]) {
    half8 av;
#pragma unroll
    for (int i = 0; i < 8; ++i) av[i] = (_Float16)areg[i];
    *(half8*)&Abuf[a_m][a_kg] = av;                // one ds_store_b128
}

#if USE_ASYNC
__device__ __forceinline__ void b_stage_async(const _Float16* __restrict__ wt2,
                                              int tap, int c0, int tid,
                                              _Float16 (*Bbuf)[KCP]) {
    gv4i* g = (gv4i*)(wt2 + ((size_t)(tap * COUT + tid)) * CIN + c0);
    lv4i* l = (lv4i*)&Bbuf[tid][0];
#pragma unroll
    for (int i = 0; i < 4; ++i)
        __builtin_amdgcn_global_load_async_to_lds_b128(g + i, l + i, 0, 0);
}
#else
__device__ __forceinline__ void b_gather(const _Float16* __restrict__ wt2,
                                         int tap, int c0, int tid, half8 breg[4]) {
    const half8* src = (const half8*)(wt2 + ((size_t)(tap * COUT + tid)) * CIN + c0);
#pragma unroll
    for (int i = 0; i < 4; ++i) breg[i] = src[i];
}
__device__ __forceinline__ void b_commit(_Float16 (*Bbuf)[KCP], int tid,
                                         const half8 breg[4]) {
    half8* dst = (half8*)&Bbuf[tid][0];
#pragma unroll
    for (int i = 0; i < 4; ++i) dst[i] = breg[i];
}
#endif

// ---------------------------------------------------------------------------
// Kernel 3: implicit-GEMM conv + fused BN + ReLU + PoT activation quant.
// grid = MTOT/MT = 1600, block = 256. Wave w: 16(M) x 128(N) -> 8 accumulators.
// Double-buffered LDS, one barrier per k-step, 8 WMMA per wave per step.
// ---------------------------------------------------------------------------
__global__ __launch_bounds__(256)
void k_conv(const float* __restrict__ x, const _Float16* __restrict__ wt2,
            const float* __restrict__ ws,
            const float* __restrict__ gamma, const float* __restrict__ beta,
            const float* __restrict__ rmean, const float* __restrict__ rvar,
            const float* __restrict__ ascale, float* __restrict__ out) {
    __shared__ __align__(16) _Float16 Als[2][MT][KCP];   // 12 KB
    __shared__ __align__(32) _Float16 Bls[2][NT][KCP];   // 48 KB

    const int tid  = threadIdx.x;
    const int bx   = blockIdx.x;
    const int lane = tid & 31;
    const int wave = tid >> 5;
    const int wm   = (wave & 3) * 16;      // wave M offset
    const int wn   = (wave >> 2) * 128;    // wave N offset

    // A staging map: thread -> (pixel, 8 consecutive channels)
    const int a_m  = tid >> 2;
    const int a_kg = (tid & 3) * 8;
    const int mg   = bx * MT + a_m;
    const int nimg = mg / IMG;
    const int hw   = mg % IMG;
    const int ph   = hw / HWD, pw = hw % HWD;
    const long xbase = (long)nimg * CIN * IMG;

    // fragment addressing (contiguous k order, k = 16h + s, same for A and B)
    const int am  = wm + (lane & 15);
    const int akb = (lane >> 4) * 16;      // A: lanes<16 K0-15, else K16-31
    const int bn  = lane & 15;
    const int bkb = (lane >> 4) * 16;      // B: lanes<16 K0-15, else K16-31

    v8f acc[8];
#pragma unroll
    for (int j = 0; j < 8; ++j)
#pragma unroll
        for (int e = 0; e < 8; ++e) acc[j][e] = 0.0f;

    // ---- prologue: stage step 0 into buffer 0 ----
    {
        float areg[8];
        a_gather(x, xbase, ph, pw, 0, 0, a_kg, areg);
        a_commit(Als[0], a_m, a_kg, areg);
#if USE_ASYNC
        b_stage_async(wt2, 0, 0, tid, Bls[0]);
        async_wait0();
#else
        half8 breg[4];
        b_gather(wt2, 0, 0, tid, breg);
        b_commit(Bls[0], tid, breg);
#endif
    }
    __syncthreads();

#pragma unroll 2
    for (int step = 0; step < NSTEP; ++step) {
        const int cur  = step & 1, nxt = cur ^ 1;
        const bool more = (step + 1) < NSTEP;
        const int tapn = (step + 1) >> 2;
        const int c0n  = ((step + 1) & 3) * KC;

        float areg[8];
#if !USE_ASYNC
        half8 breg[4];
#endif
        if (more) {
#if USE_ASYNC
            b_stage_async(wt2, tapn, c0n, tid, Bls[nxt]);   // ASYNCcnt, direct to LDS
#else
            b_gather(wt2, tapn, c0n, tid, breg);
#endif
            a_gather(x, xbase, ph, pw, tapn, c0n, a_kg, areg);
        }

        // ---- compute on current buffer: 1 A frag + 8 B frags + 8 WMMA ----
        const half8 alo = *(const half8*)&Als[cur][am][akb];
        const half8 ahi = *(const half8*)&Als[cur][am][akb + 8];
        v16h a;
#pragma unroll
        for (int i = 0; i < 8; ++i) { a[i] = alo[i]; a[8 + i] = ahi[i]; }
#pragma unroll
        for (int j = 0; j < 8; ++j) {
            const v16h b = *(const v16h*)&Bls[cur][wn + j * 16 + bn][bkb];
            acc[j] = __builtin_amdgcn_wmma_f32_16x16x32_f16(
                false, a, false, b, (short)0, acc[j], false, false);
        }

        if (more) {
            a_commit(Als[nxt], a_m, a_kg, areg);
#if !USE_ASYNC
            b_commit(Bls[nxt], tid, breg);
#endif
        }
#if USE_ASYNC
        async_wait0();
#endif
        __syncthreads();
    }

    // ---- fused epilogue: *w_scale, BN(running stats), ReLU, PoT quant ----
    const float wsc = ws[1];
    const float sp  = exp2f(ceilf(log2f(ascale[0])));
    const float isp = 1.0f / sp;
    const int nn = lane & 15;
    const int mh = (lane >> 4) * 8;     // C/D layout: VGPR r -> M=r (+8 hi lanes)

#pragma unroll
    for (int j = 0; j < 8; ++j) {
        const int ch = wn + j * 16 + nn;
        const float ivs = rsqrtf(rvar[ch] + 1e-5f);
        const float bsc = gamma[ch] * ivs;
        const float bsh = beta[ch] - rmean[ch] * bsc;
#pragma unroll
        for (int r2 = 0; r2 < 8; ++r2) {
            const int m   = bx * MT + wm + mh + r2;
            const int ni  = m / IMG;
            const int hw2 = m % IMG;
            float y = acc[j][r2] * wsc * bsc + bsh;
            y = fmaxf(y, 0.0f);
            const float q = fminf(fmaxf(rintf(y * isp), 0.0f), 63.0f) * sp;
            out[((size_t)(ni * COUT + ch)) * IMG + hw2] = q;
        }
    }
}

// ---------------------------------------------------------------------------
extern "C" void kernel_launch(void* const* d_in, const int* in_sizes, int n_in,
                              void* d_out, int out_size, void* d_ws, size_t ws_size,
                              hipStream_t stream) {
    const float* x      = (const float*)d_in[0];
    const float* w      = (const float*)d_in[1];
    const float* gamma  = (const float*)d_in[2];
    const float* beta   = (const float*)d_in[3];
    const float* rmean  = (const float*)d_in[4];
    const float* rvar   = (const float*)d_in[5];
    const float* ascale = (const float*)d_in[6];
    float* out = (float*)d_out;

    unsigned* wsu = (unsigned*)d_ws;                 // [0] absmax bits
    float*    wsf = (float*)d_ws;                    // [1] w_scale
    _Float16* wt2 = (_Float16*)((char*)d_ws + 256);  // 576 KB f16 weights [tap][O][C]

    k_init<<<1, 1, 0, stream>>>(wsu);
    k_absmax<<<288, 256, 0, stream>>>(w, wsu);
    k_wquant<<<(WNELE + 255) / 256, 256, 0, stream>>>(w, wsu, wsf, wt2);
    k_conv<<<MTOT / MT, 256, 0, stream>>>(x, wt2, wsf, gamma, beta, rmean, rvar,
                                          ascale, out);
}
